// SelfAttention__gen062_65481071407245
// MI455X (gfx1250) — compile-verified, matrix operand order checked by reading, not run on hardware
//
#include <hip/hip_runtime.h>
#include <hip/hip_bf16.h>

// ---------------------------------------------------------------------------
// Self-attention B=4, S=2048, D=1024 (fp32 ref) on gfx1250 / MI455X.
// bf16 WMMA (v_wmma_f32_16x16x32_bf16) with fp32 accumulation for all GEMMs.
// GEMM core: 128x128 workgroup tile, BK=64, double-buffered LDS fed by
// GLOBAL_LOAD_ASYNC_TO_LDS_B128 (ASYNCcnt-pipelined) when available.
// ---------------------------------------------------------------------------

typedef __attribute__((ext_vector_type(16))) __bf16 v16bf;
typedef __attribute__((ext_vector_type(8)))  float  v8f;
typedef int v4i __attribute__((vector_size(16)));   // matches builtin param type

constexpr int B_ = 4, S_ = 2048, D_ = 1024;
constexpr int BM = 128, BN = 128, BK = 64;

enum { EPI_BF16 = 0, EPI_BF16_T = 1, EPI_F32_MASK = 2, EPI_F32 = 3 };

#if defined(__AMDGCN__) && __has_builtin(__builtin_amdgcn_global_load_async_to_lds_b128)
#define HAVE_ASYNC_LDS 1
#else
#define HAVE_ASYNC_LDS 0
#endif

// Copy 16 bytes global -> LDS. Async (direct-to-LDS, ASYNCcnt) when supported.
__device__ inline void cp16_g2l(const __bf16* __restrict__ g, __bf16* __restrict__ l) {
#if HAVE_ASYNC_LDS
  typedef __attribute__((address_space(1))) v4i GV4;
  typedef __attribute__((address_space(3))) v4i LV4;
  // Generic LDS address: low 32 bits are the as3 offset (ISA aperture rule).
  __builtin_amdgcn_global_load_async_to_lds_b128(
      (GV4*)(uintptr_t)g, (LV4*)(uint32_t)(uintptr_t)l, /*offset=*/0, /*cpol=*/0);
#else
  *(uint4*)l = *(const uint4*)g;
#endif
}

template<int N>
__device__ inline void wait_async_le() {
#if HAVE_ASYNC_LDS
#if __has_builtin(__builtin_amdgcn_s_wait_asynccnt)
  __builtin_amdgcn_s_wait_asynccnt(N);
#else
  asm volatile("s_wait_asynccnt %0" :: "i"(N) : "memory");
#endif
#endif
}

// Load a 16x32 bf16 WMMA operand fragment from a row-major LDS tile with row
// stride BK elems. CDNA5 16-bit operand lane layout:
//   lanes 0-15 : row = lane,    K = [0..7] then [16..23]
//   lanes 16-31: row = lane-16, K = [8..15] then [24..31]
__device__ inline v16bf frag_ld(const __bf16* base, int lane) {
  const int r  = lane & 15;
  const int ko = (lane >> 4) << 3;                 // 0 or 8
  const __bf16* p = base + r * BK + ko;
  union { uint4 u[2]; v16bf v; } f;
  f.u[0] = *(const uint4*)(p);                     // K = ko .. ko+7
  f.u[1] = *(const uint4*)(p + 16);                // K = ko+16 .. ko+23
  return f.v;
}

// C(MxN) = A(MxK, lda) * Bt(NxK, ldb)^T, bf16 in / fp32 accum.
// 8 waves; wave (wm,wn) in a 2x4 grid owns a 64x32 sub-tile = 4x2 accumulators.
// Double-buffered LDS stages of 128x64, async-copy pipelined.
template<int EPI>
__global__ __launch_bounds__(256)
void gemm_bf16_wmma(const __bf16* __restrict__ A, const __bf16* __restrict__ Bt,
                    void* __restrict__ Cv, const float* __restrict__ mask,
                    int M, int N, int K, int lda, int ldb, int ldc,
                    long sA, long sB, long sC, long sMask, float scale)
{
  __shared__ __align__(16) __bf16 As[2][BM * BK];   // 2 x 16KB
  __shared__ __align__(16) __bf16 Bs[2][BN * BK];   // 2 x 16KB

  const int z = blockIdx.z;
  A  += z * sA;
  Bt += z * sB;

  const int tid  = threadIdx.x;
  const int lane = tid & 31;
  const int wave = tid >> 5;        // 0..7 (wave32)
  const int wm   = wave >> 2;       // 0..1 -> M offset wm*64
  const int wn   = wave & 3;        // 0..3 -> N offset wn*32
  const int bm   = blockIdx.y * BM;
  const int bn   = blockIdx.x * BN;

  // Per-thread stage copy: 128x64 bf16 tile = 1024 x 16B chunks; 4 per thread
  // per matrix (8 async ops per thread per stage).
  auto issue_stage = [&](int kt, int buf) {
    const int k0 = kt * BK;
#pragma unroll
    for (int p = 0; p < 4; ++p) {
      const int c   = tid + p * 256;      // 0..1023
      const int row = c >> 3;             // 0..127   (8 chunks per 64-elem row)
      const int col = (c & 7) * 8;        // element offset within row
      cp16_g2l(A  + (long)(bm + row) * lda + k0 + col, &As[buf][row * BK + col]);
      cp16_g2l(Bt + (long)(bn + row) * ldb + k0 + col, &Bs[buf][row * BK + col]);
    }
  };

  v8f acc[4][2];
#pragma unroll
  for (int i = 0; i < 4; ++i)
#pragma unroll
    for (int j = 0; j < 2; ++j)
#pragma unroll
      for (int e = 0; e < 8; ++e) acc[i][j][e] = 0.0f;

  const int nst = K / BK;
  issue_stage(0, 0);

  for (int kt = 0; kt < nst; ++kt) {
    const int cur = kt & 1;
    if (kt + 1 < nst) {
      issue_stage(kt + 1, cur ^ 1);      // prefetch next stage into other buffer
      wait_async_le<8>();                // current stage's 8 ops done (in-order)
    } else {
      wait_async_le<0>();
    }
    __syncthreads();

    const __bf16* bufA = As[cur];
    const __bf16* bufB = Bs[cur];
#pragma unroll
    for (int ks = 0; ks < 2; ++ks) {     // two 32-wide K-steps per stage
      v16bf af[4], bfr[2];
#pragma unroll
      for (int i = 0; i < 4; ++i)
        af[i]  = frag_ld(bufA + (wm * 64 + i * 16) * BK + ks * 32, lane);
#pragma unroll
      for (int j = 0; j < 2; ++j)
        bfr[j] = frag_ld(bufB + (wn * 32 + j * 16) * BK + ks * 32, lane);
#pragma unroll
      for (int i = 0; i < 4; ++i)
#pragma unroll
        for (int j = 0; j < 2; ++j)
          acc[i][j] = __builtin_amdgcn_wmma_f32_16x16x32_bf16(
              /*neg_a=*/false, af[i], /*neg_b=*/false, bfr[j],
              /*c_mod=*/(short)0, acc[i][j], /*reuse_a=*/false, /*reuse_b=*/false);
    }
    __syncthreads();                     // buffer safe for reuse 2 stages later
  }

  // Epilogue. C/D layout: VGPR g, lanes 0-15 -> (M=g, N=lane); 16-31 -> (M=8+g, N=lane-16).
  const int r0   = bm + wm * 64;
  const int c0   = bn + wn * 32;
  const int colL = lane & 15;
  const int rOff = (lane >> 4) * 8;
#pragma unroll
  for (int i = 0; i < 4; ++i)
#pragma unroll
    for (int j = 0; j < 2; ++j) {
      const int col     = c0 + j * 16 + colL;
      const int rowBase = r0 + i * 16 + rOff;
#pragma unroll
      for (int g = 0; g < 8; ++g) {
        const int row = rowBase + g;
        const float v = acc[i][j][g];
        if constexpr (EPI == EPI_BF16) {
          ((__bf16*)Cv)[z * sC + (long)row * ldc + col] = (__bf16)v;
        } else if constexpr (EPI == EPI_BF16_T) {
          // global row in [0, B*S): store V transposed as vT[b][col][s]
          const int b = row >> 11;            // / S_
          const int s = row & (S_ - 1);
          ((__bf16*)Cv)[((long)b * D_ + col) * (long)S_ + s] = (__bf16)v;
        } else if constexpr (EPI == EPI_F32_MASK) {
          const float m = mask[z * sMask + row];
          ((float*)Cv)[z * sC + (long)row * ldc + col] = v * scale * m;
        } else {
          ((float*)Cv)[z * sC + (long)row * ldc + col] = v;
        }
      }
    }
}

// Row softmax over 2048 fp32 values; writes bf16 probabilities in place within
// the same row's storage (bf16 row stride = 2*S_ elements).
__global__ __launch_bounds__(256)
void softmax_rows(float* __restrict__ scores)
{
  const long row = blockIdx.x;
  float*  p = scores + row * (long)S_;
  __bf16* o = (__bf16*)p;                    // overlays this row's own bytes
  const int tid = threadIdx.x;

  const float4 a = ((const float4*)p)[tid];
  const float4 b = ((const float4*)p)[tid + 256];
  float v[8] = {a.x, a.y, a.z, a.w, b.x, b.y, b.z, b.w};

  float mx = v[0];
#pragma unroll
  for (int i = 1; i < 8; ++i) mx = fmaxf(mx, v[i]);

  __shared__ float red[256];
  red[tid] = mx;
  __syncthreads();
  for (int s = 128; s > 0; s >>= 1) {
    if (tid < s) red[tid] = fmaxf(red[tid], red[tid + s]);
    __syncthreads();
  }
  mx = red[0];
  __syncthreads();

  float e[8], sum = 0.0f;
#pragma unroll
  for (int i = 0; i < 8; ++i) { e[i] = __expf(v[i] - mx); sum += e[i]; }
  red[tid] = sum;
  __syncthreads();
  for (int s = 128; s > 0; s >>= 1) {
    if (tid < s) red[tid] += red[tid + s];
    __syncthreads();
  }
  const float inv = 1.0f / red[0];

#pragma unroll
  for (int i = 0; i < 4; ++i) o[4 * tid + i]         = (__bf16)(e[i] * inv);
#pragma unroll
  for (int i = 0; i < 4; ++i) o[4 * (tid + 256) + i] = (__bf16)(e[4 + i] * inv);
}

__global__ __launch_bounds__(256)
void cvt_f32_bf16(const float* __restrict__ in, __bf16* __restrict__ out, long n)
{
  const long i = ((long)blockIdx.x * 256 + threadIdx.x) * 4;
  if (i < n) {
    const float4 f = *(const float4*)(in + i);
    out[i + 0] = (__bf16)f.x;
    out[i + 1] = (__bf16)f.y;
    out[i + 2] = (__bf16)f.z;
    out[i + 3] = (__bf16)f.w;
  }
}

extern "C" void kernel_launch(void* const* d_in, const int* in_sizes, int n_in,
                              void* d_out, int out_size, void* d_ws, size_t ws_size,
                              hipStream_t stream)
{
  (void)in_sizes; (void)n_in; (void)out_size; (void)ws_size;
  const float* x    = (const float*)d_in[0];
  const float* Wq   = (const float*)d_in[1];
  const float* Wk   = (const float*)d_in[2];
  const float* Wv   = (const float*)d_in[3];
  const float* qmsk = (const float*)d_in[4];
  float* out = (float*)d_out;

  // Workspace: xb 16M | wq/wk/wv 2M each | qb 16M | kb 16M | vT 16M | scores 64M
  char* ws = (char*)d_ws;
  const long XN = (long)B_ * S_ * D_;          // 8388608
  const long WN = (long)D_ * D_;               // 1048576
  __bf16* xb  = (__bf16*)ws;  ws += XN * 2;
  __bf16* wqb = (__bf16*)ws;  ws += WN * 2;
  __bf16* wkb = (__bf16*)ws;  ws += WN * 2;
  __bf16* wvb = (__bf16*)ws;  ws += WN * 2;
  __bf16* qb  = (__bf16*)ws;  ws += XN * 2;
  __bf16* kb  = (__bf16*)ws;  ws += XN * 2;
  __bf16* vT  = (__bf16*)ws;  ws += XN * 2;
  float*  sc  = (float*)ws;                    // B*S*S fp32 (P bf16 overlaid in-row)

  const float scale = 0.03125f;                // 1/sqrt(1024)

  // 1) fp32 -> bf16 conversions
  cvt_f32_bf16<<<dim3((unsigned)(XN / 1024)), 256, 0, stream>>>(x,  xb,  XN);
  cvt_f32_bf16<<<dim3((unsigned)(WN / 1024)), 256, 0, stream>>>(Wq, wqb, WN);
  cvt_f32_bf16<<<dim3((unsigned)(WN / 1024)), 256, 0, stream>>>(Wk, wkb, WN);
  cvt_f32_bf16<<<dim3((unsigned)(WN / 1024)), 256, 0, stream>>>(Wv, wvb, WN);

  // 2) Projections: C(8192x1024) = xb @ W^T ; W row-major (out,in) == Bt(NxK)
  const dim3 gProj(D_ / BN, (B_ * S_) / BM, 1);
  gemm_bf16_wmma<EPI_BF16><<<gProj, 256, 0, stream>>>(
      xb, wqb, qb, nullptr, B_ * S_, D_, D_, D_, D_, D_, 0, 0, 0, 0, 1.0f);
  gemm_bf16_wmma<EPI_BF16><<<gProj, 256, 0, stream>>>(
      xb, wkb, kb, nullptr, B_ * S_, D_, D_, D_, D_, D_, 0, 0, 0, 0, 1.0f);
  gemm_bf16_wmma<EPI_BF16_T><<<gProj, 256, 0, stream>>>(
      xb, wvb, vT, nullptr, B_ * S_, D_, D_, D_, D_, D_, 0, 0, 0, 0, 1.0f);

  // 3) Scores: per batch, S(2048x2048) = scale * mask[row] * (Q @ K^T)
  const dim3 gScore(S_ / BN, S_ / BM, B_);
  gemm_bf16_wmma<EPI_F32_MASK><<<gScore, 256, 0, stream>>>(
      qb, kb, sc, qmsk, S_, S_, D_, D_, D_, S_,
      (long)S_ * D_, (long)S_ * D_, (long)S_ * S_, (long)S_, scale);

  // 4) Row softmax -> bf16 P stored in-row (lda = 2*S_ bf16 elements)
  softmax_rows<<<dim3(B_ * S_), 256, 0, stream>>>(sc);

  // 5) Output: per batch, O(2048x1024) = P(2048x2048 bf16, lda=4096) @ vT^T
  const dim3 gOut(D_ / BN, S_ / BM, B_);
  gemm_bf16_wmma<EPI_F32><<<gOut, 256, 0, stream>>>(
      (const __bf16*)sc, vT, out, nullptr, S_, D_, S_, 2 * S_, S_, D_,
      (long)S_ * S_ * 2, (long)D_ * S_, (long)S_ * D_, 0, 1.0f);
}
